// PointNet2_3049426780799
// MI455X (gfx1250) — matrix operand order repaired and checked by reading, not compiled, not run on hardware
//
#include <hip/hip_runtime.h>

// ---------------------------------------------------------------------------
// PointNet++ (MSG) forward for MI455X / gfx1250.
// Compute-dominant path (grouped MLP GEMMs) runs on v_wmma_f32_16x16x32_f16
// with 32x32 register macro-tiles per wave (4 WMMAs per 2 A + 2 B fragment
// loads) so the LDS interface is not the limiter. SA stages are fully fused
// per group in LDS (ball query -> gather -> 3-layer MLP -> maxpool).
// The K=16 edge case (1x2 tiles) is hoisted out of the k-loop so the
// steady-state inner loop has no EXEC manipulation around WMMA.
// ---------------------------------------------------------------------------

typedef _Float16 v8h  __attribute__((ext_vector_type(8)));
typedef _Float16 v16h __attribute__((ext_vector_type(16)));
typedef float    v8f  __attribute__((ext_vector_type(8)));
union HV { v16h v; v8h h[2]; };

#define INVSTD 0.99999500003749968f   // 1/sqrt(1+1e-5)
#define WMMA_F16(Af, Bf, Cf) \
  __builtin_amdgcn_wmma_f32_16x16x32_f16(false, (Af), false, (Bf), (short)0, (Cf), false, false)

// ---------------------------------------------------------------------------
// 0) transpose (B,6,N) -> pts (B,N,3), nrm (B,N,3)
// ---------------------------------------------------------------------------
__global__ void transpose_kernel(const float* __restrict__ xyz,
                                 float* __restrict__ pts, float* __restrict__ nrm,
                                 int B, int N) {
  int t = blockIdx.x * blockDim.x + threadIdx.x;
  if (t >= B * N) return;
  int b = t / N, n = t - b * N;
  const float* src = xyz + (size_t)b * 6 * N;
  for (int d = 0; d < 3; d++) {
    pts[(size_t)t * 3 + d] = src[(size_t)d * N + n];
    nrm[(size_t)t * 3 + d] = src[(size_t)(d + 3) * N + n];
  }
}

// ---------------------------------------------------------------------------
// 1) Farthest-point sampling. One block per batch; coords + min-dist in LDS.
//    Emits the *previous* farthest each step (scan semantics: out[0] = 0).
// ---------------------------------------------------------------------------
__global__ void __launch_bounds__(512) fps_kernel(const float* __restrict__ xyz,
                                                  int N, int npoint,
                                                  int* __restrict__ out) {
  extern __shared__ float sm[];
  float* xs = sm;
  float* ys = sm + N;
  float* zs = sm + 2 * N;
  float* dist = sm + 3 * N;
  __shared__ float rv[16];
  __shared__ int   ri[16];
  __shared__ int   far_s;

  const int b = blockIdx.x, tid = threadIdx.x;
  const int lane = tid & 31, wid = tid >> 5, nwv = blockDim.x >> 5;

  for (int j = tid; j < N; j += blockDim.x) {
    const float* p = xyz + ((size_t)b * N + j) * 3;
    xs[j] = p[0]; ys[j] = p[1]; zs[j] = p[2];
    dist[j] = 1e10f;
  }
  __syncthreads();

  int farthest = 0;
  for (int i = 0; i < npoint; i++) {
    if (tid == 0) out[b * npoint + i] = farthest;
    float cx = xs[farthest], cy = ys[farthest], cz = zs[farthest];
    float bestv = -1.f; int besti = 0;
    for (int j = tid; j < N; j += blockDim.x) {
      float dx = xs[j] - cx, dy = ys[j] - cy, dz = zs[j] - cz;
      float d = dx * dx + dy * dy + dz * dz;
      float nd = fminf(dist[j], d);
      dist[j] = nd;                       // thread-private slots: no race
      if (nd > bestv) { bestv = nd; besti = j; }
    }
    for (int off = 16; off > 0; off >>= 1) {
      float ov = __shfl_down(bestv, off, 32);
      int   oi = __shfl_down(besti, off, 32);
      if (ov > bestv || (ov == bestv && oi < besti)) { bestv = ov; besti = oi; }
    }
    if (lane == 0) { rv[wid] = bestv; ri[wid] = besti; }
    __syncthreads();
    if (tid == 0) {
      float bv = rv[0]; int bi = ri[0];
      for (int w = 1; w < nwv; w++)
        if (rv[w] > bv || (rv[w] == bv && ri[w] < bi)) { bv = rv[w]; bi = ri[w]; }
      far_s = bi;
    }
    __syncthreads();
    farthest = far_s;
    __syncthreads();
  }
}

__global__ void gather_kernel(const float* __restrict__ pts, const int* __restrict__ idx,
                              float* __restrict__ out, int N, int S, int B) {
  int t = blockIdx.x * blockDim.x + threadIdx.x;
  if (t >= B * S) return;
  int b = t / S, p = idx[t];
  for (int d = 0; d < 3; d++)
    out[(size_t)t * 3 + d] = pts[((size_t)b * N + p) * 3 + d];
}

// ---------------------------------------------------------------------------
// 2) One WMMA MLP layer fully in LDS, 32x32 macro-tiles per wave.
//    in: K x Din (f16), W staged to LDS as [Dout][Din] (k-contiguous),
//    out: K x Dout (f16) post ReLU(acc*g'+b).
//    A/B k order used here (contiguous, not AMD table order): A lanes 0-15 -> K 0..15,
//    lanes 16-31 -> K 16..31. B-frag (32x16): lanes 0-15 col n,
//    K 0..15; lanes 16-31 col n, K 16..31.
// ---------------------------------------------------------------------------
__device__ __forceinline__ void mlp_layer(const _Float16* __restrict__ in,
                                          _Float16* __restrict__ outb,
                                          _Float16* __restrict__ Wlds,
                                          const float* __restrict__ Wg,
                                          const float* __restrict__ gg,
                                          const float* __restrict__ bg,
                                          int K, int Din, int DinReal, int Dout) {
  const int tid = threadIdx.x, lane = tid & 31, wid = tid >> 5;
  const int nwaves = blockDim.x >> 5;

  for (int e = tid; e < Dout * Din; e += blockDim.x) {
    int n = e / Din, k = e - n * Din;
    Wlds[e] = (k < DinReal) ? (_Float16)Wg[n * DinReal + k] : (_Float16)0.f;
  }
  __syncthreads();

  const int tilesN2 = Dout >> 5;           // 32-wide macro columns (Dout % 32 == 0)
  const int macrosM = (K + 31) >> 5;       // 32-tall macro rows (K=16 -> half macro)
  const int total   = macrosM * tilesN2;
  const int klo  = (lane & 16) ? 8 : 0;
  const int koff = (lane & 16) ? 16 : 0;
  const int l15  = lane & 15;

  for (int mt = wid; mt < total; mt += nwaves) {
    int tm2 = mt / tilesN2, tn2 = mt - tm2 * tilesN2;
    int row0 = (tm2 << 5) + l15;
    int col0 = (tn2 << 5) + l15;
    bool hasM1 = ((tm2 << 5) + 31) < K;    // wave-uniform: 2nd 16-row tile exists
    const _Float16* ar0 = in + row0 * Din;
    const _Float16* ar1 = ar0 + 16 * Din;
    const _Float16* br0 = Wlds + col0 * Din;
    const _Float16* br1 = br0 + 16 * Din;
    v8f acc00 = {}, acc01 = {}, acc10 = {}, acc11 = {};
    if (hasM1) {
      // steady-state 2x2 macro-tile: no EXEC games in the inner loop
      for (int kk = 0; kk < Din; kk += 32) {
        HV a0, a1;
        a0.h[0] = *(const v8h*)(ar0 + kk + koff);
        a0.h[1] = *(const v8h*)(ar0 + kk + koff + 8);
        a1.h[0] = *(const v8h*)(ar1 + kk + koff);
        a1.h[1] = *(const v8h*)(ar1 + kk + koff + 8);
        v16h b0 = *(const v16h*)(br0 + kk + koff);
        v16h b1 = *(const v16h*)(br1 + kk + koff);
        acc00 = WMMA_F16(a0.v, b0, acc00);
        acc01 = WMMA_F16(a0.v, b1, acc01);
        acc10 = WMMA_F16(a1.v, b0, acc10);
        acc11 = WMMA_F16(a1.v, b1, acc11);
      }
    } else {
      // K=16 tail: 1x2 tiles
      for (int kk = 0; kk < Din; kk += 32) {
        HV a0;
        a0.h[0] = *(const v8h*)(ar0 + kk + koff);
        a0.h[1] = *(const v8h*)(ar0 + kk + koff + 8);
        v16h b0 = *(const v16h*)(br0 + kk + koff);
        v16h b1 = *(const v16h*)(br1 + kk + koff);
        acc00 = WMMA_F16(a0.v, b0, acc00);
        acc01 = WMMA_F16(a0.v, b1, acc01);
      }
    }
    float gs0 = gg[col0] * INVSTD,      bv0 = bg[col0];
    float gs1 = gg[col0 + 16] * INVSTD, bv1 = bg[col0 + 16];
    int rb0 = (tm2 << 5) + ((lane & 16) ? 8 : 0);
#pragma unroll
    for (int j = 0; j < 8; j++) {
      float v0 = acc00[j] * gs0 + bv0; v0 = v0 > 0.f ? v0 : 0.f;
      float v1 = acc01[j] * gs1 + bv1; v1 = v1 > 0.f ? v1 : 0.f;
      outb[(rb0 + j) * Dout + col0]      = (_Float16)v0;
      outb[(rb0 + j) * Dout + col0 + 16] = (_Float16)v1;
    }
    if (hasM1) {
      int rb1 = rb0 + 16;
#pragma unroll
      for (int j = 0; j < 8; j++) {
        float v0 = acc10[j] * gs0 + bv0; v0 = v0 > 0.f ? v0 : 0.f;
        float v1 = acc11[j] * gs1 + bv1; v1 = v1 > 0.f ? v1 : 0.f;
        outb[(rb1 + j) * Dout + col0]      = (_Float16)v0;
        outb[(rb1 + j) * Dout + col0 + 16] = (_Float16)v1;
      }
    }
  }
  __syncthreads();
}

// ---------------------------------------------------------------------------
// 3) Fused SA branch: one workgroup per (b,s) group.
//    ball query -> gather (f16 into LDS) -> 3 WMMA MLP layers -> maxpool.
// ---------------------------------------------------------------------------
__global__ void __launch_bounds__(256) sa_branch_kernel(
    const float* __restrict__ pts, const float* __restrict__ feats,
    const float* __restrict__ nxyz, int N, int S, int K,
    int Cpts, int Cin0, int Cin0Pad, float r2,
    const float* W0, const float* g0, const float* b0, int C1,
    const float* W1, const float* g1, const float* b1, int C2,
    const float* W2, const float* g2, const float* b2, int C3,
    int Xcap, int Ycap, float* __restrict__ out, int outStride, int outOff) {
  extern __shared__ char smem[];
  const int bs = blockIdx.x, b = bs / S;
  const int tid = threadIdx.x;
  const int nw32 = N >> 5;

  int*      idxs = (int*)smem;
  unsigned* mask = (unsigned*)(idxs + K);
  size_t hdr = ((size_t)(K * 4 + nw32 * 4) + 63) & ~(size_t)63;
  _Float16* X  = (_Float16*)(smem + hdr);
  _Float16* Y  = X + (size_t)K * Xcap;
  _Float16* Wl = Y + (size_t)K * Ycap;

  const float* cp = nxyz + (size_t)bs * 3;
  const float cx = cp[0], cy = cp[1], cz = cp[2];

  // --- ball query: bitmask of in-radius points (ascending index order) ---
  for (int w = tid; w < nw32; w += blockDim.x) {
    unsigned m = 0;
    int base = w << 5;
    for (int t = 0; t < 32; t++) {
      const float* p = pts + ((size_t)b * N + base + t) * 3;
      float dx = p[0] - cx, dy = p[1] - cy, dz = p[2] - cz;
      if (dx * dx + dy * dy + dz * dz <= r2) m |= (1u << t);
    }
    mask[w] = m;
  }
  __syncthreads();
  if (tid == 0) {
    int cnt = 0;
    for (int w = 0; w < nw32 && cnt < K; w++) {
      unsigned m = mask[w];
      while (m && cnt < K) {
        int bit = __ffs((int)m) - 1;
        idxs[cnt++] = (w << 5) + bit;
        m &= m - 1;
      }
    }
    int f0 = cnt ? idxs[0] : 0;           // centroid always matches -> cnt>=1
    while (cnt < K) idxs[cnt++] = f0;
  }
  __syncthreads();

  // --- gather group features into LDS (f16), zero-padded to Cin0Pad ---
  for (int e = tid; e < K * Cin0Pad; e += blockDim.x) {
    int k = e / Cin0Pad, c = e - k * Cin0Pad;
    float v = 0.f;
    int p = idxs[k];
    if (c < Cpts) {
      v = feats[((size_t)b * N + p) * Cpts + c];
    } else if (c < Cin0) {
      int d = c - Cpts;
      const float* pp = pts + ((size_t)b * N + p) * 3;
      float cen = (d == 0) ? cx : ((d == 1) ? cy : cz);
      v = pp[d] - cen;
    }
    X[e] = (_Float16)v;
  }
  // (barrier happens inside mlp_layer after W staging)

  mlp_layer(X, Y, Wl, W0, g0, b0, K, Cin0Pad, Cin0, C1);
  mlp_layer(Y, X, Wl, W1, g1, b1, K, C1, C1, C2);
  mlp_layer(X, Y, Wl, W2, g2, b2, K, C2, C2, C3);

  // --- maxpool over K (post-ReLU => >= 0) ---
  for (int c = tid; c < C3; c += blockDim.x) {
    float m = 0.f;
    for (int k = 0; k < K; k++) {
      float v = (float)Y[k * C3 + c];
      m = v > m ? v : m;
    }
    out[(size_t)bs * outStride + outOff + c] = m;
  }
}

// ---------------------------------------------------------------------------
// 4) SA3 / global-memory WMMA GEMM, 32x32 macro-tile per wave:
//    Y = relu((A @ Bw^T)*g' + b), f16 out.
//    A: M x Kpad (f16, row-major), Bw: N x Kpad (f16, k-contiguous, padded).
//    M % 32 == 0 and Nn % 32 == 0 for all call sites.
// ---------------------------------------------------------------------------
__global__ void __launch_bounds__(256) wmma_gemm_kernel(
    const _Float16* __restrict__ A, const _Float16* __restrict__ Bw,
    const float* __restrict__ g, const float* __restrict__ bb,
    _Float16* __restrict__ Y, int M, int Kpad, int Nn) {
  const int lane = threadIdx.x & 31, wid = threadIdx.x >> 5;
  const int tilesN2 = Nn >> 5;
  const int total   = (M >> 5) * tilesN2;
  int mt = blockIdx.x * (blockDim.x >> 5) + wid;
  if (mt >= total) return;                 // wave-uniform exit
  int tm2 = mt / tilesN2, tn2 = mt - tm2 * tilesN2;
  const int klo  = (lane & 16) ? 8 : 0;
  const int koff = (lane & 16) ? 16 : 0;
  const int l15  = lane & 15;
  int row0 = (tm2 << 5) + l15;
  int col0 = (tn2 << 5) + l15;
  const _Float16* ar0 = A  + (size_t)row0 * Kpad;
  const _Float16* ar1 = ar0 + (size_t)16 * Kpad;
  const _Float16* br0 = Bw + (size_t)col0 * Kpad;
  const _Float16* br1 = br0 + (size_t)16 * Kpad;
  v8f acc00 = {}, acc01 = {}, acc10 = {}, acc11 = {};
  for (int kk = 0; kk < Kpad; kk += 32) {
    if (kk + 256 < Kpad) {                 // stream ahead via global_prefetch
      __builtin_prefetch(ar0 + kk + 256, 0, 0);
      __builtin_prefetch(br0 + kk + 256, 0, 0);
    }
    HV a0, a1;
    a0.h[0] = *(const v8h*)(ar0 + kk + koff);
    a0.h[1] = *(const v8h*)(ar0 + kk + koff + 8);
    a1.h[0] = *(const v8h*)(ar1 + kk + koff);
    a1.h[1] = *(const v8h*)(ar1 + kk + koff + 8);
    v16h b0 = *(const v16h*)(br0 + kk + koff);
    v16h b1 = *(const v16h*)(br1 + kk + koff);
    acc00 = WMMA_F16(a0.v, b0, acc00);
    acc01 = WMMA_F16(a0.v, b1, acc01);
    acc10 = WMMA_F16(a1.v, b0, acc10);
    acc11 = WMMA_F16(a1.v, b1, acc11);
  }
  float gs0 = g[col0] * INVSTD,      bv0 = bb[col0];
  float gs1 = g[col0 + 16] * INVSTD, bv1 = bb[col0 + 16];
  int rb0 = (tm2 << 5) + ((lane & 16) ? 8 : 0);
  int rb1 = rb0 + 16;
#pragma unroll
  for (int j = 0; j < 8; j++) {
    float v00 = acc00[j] * gs0 + bv0; v00 = v00 > 0.f ? v00 : 0.f;
    float v01 = acc01[j] * gs1 + bv1; v01 = v01 > 0.f ? v01 : 0.f;
    float v10 = acc10[j] * gs0 + bv0; v10 = v10 > 0.f ? v10 : 0.f;
    float v11 = acc11[j] * gs1 + bv1; v11 = v11 > 0.f ? v11 : 0.f;
    Y[(size_t)(rb0 + j) * Nn + col0]      = (_Float16)v00;
    Y[(size_t)(rb0 + j) * Nn + col0 + 16] = (_Float16)v01;
    Y[(size_t)(rb1 + j) * Nn + col0]      = (_Float16)v10;
    Y[(size_t)(rb1 + j) * Nn + col0 + 16] = (_Float16)v11;
  }
}

// ---------------------------------------------------------------------------
// 5) Misc small kernels
// ---------------------------------------------------------------------------
__global__ void wpad_kernel(const float* __restrict__ W, _Float16* __restrict__ out,
                            int Np, int Kin, int Kpad) {
  int e = blockIdx.x * blockDim.x + threadIdx.x;
  if (e >= Np * Kpad) return;
  int n = e / Kpad, k = e - n * Kpad;
  out[e] = (k < Kin) ? (_Float16)W[(size_t)n * Kin + k] : (_Float16)0.f;
}

__global__ void concat3_kernel(const float* __restrict__ l2xyz, const float* __restrict__ l2p,
                               _Float16* __restrict__ X, int M) {
  int e = blockIdx.x * blockDim.x + threadIdx.x;
  if (e >= M * 672) return;
  int row = e / 672, c = e - row * 672;
  float v = 0.f;
  if (c < 3) v = l2xyz[(size_t)row * 3 + c];
  else if (c < 643) v = l2p[(size_t)row * 640 + (c - 3)];
  X[e] = (_Float16)v;
}

__global__ void maxpool3_kernel(const _Float16* __restrict__ Y, float* __restrict__ l3,
                                float* __restrict__ dout2, int B) {
  int t = blockIdx.x * blockDim.x + threadIdx.x;
  if (t >= B * 1024) return;
  int b = t / 1024, c = t - b * 1024;
  float m = 0.f;
  for (int s = 0; s < 128; s++) {
    float v = (float)Y[((size_t)b * 128 + s) * 1024 + c];
    m = v > m ? v : m;
  }
  l3[t] = m;
  dout2[t] = m;          // second tuple output: l3_p (B,1024,1) flat
}

__global__ void fc_kernel(const float* __restrict__ x, const float* __restrict__ W,
                          const float* __restrict__ fb, const float* __restrict__ g,
                          const float* __restrict__ bb, float* __restrict__ y,
                          int Bn, int Cin, int Cout, int mode) {
  int t = blockIdx.x * blockDim.x + threadIdx.x;
  if (t >= Bn * Cout) return;
  int b = t / Cout, o = t - b * Cout;
  const float* xr = x + (size_t)b * Cin;
  const float* wr = W + (size_t)o * Cin;
  float acc = 0.f;
  for (int c = 0; c < Cin; c++) acc = fmaf(xr[c], wr[c], acc);
  acc += fb[o];
  if (mode) {
    acc = acc * (g[o] * INVSTD) + bb[o];
    acc = acc > 0.f ? acc : 0.f;
  }
  y[t] = acc;
}

__global__ void logsoftmax_kernel(const float* __restrict__ in, float* __restrict__ out,
                                  int B, int C) {
  int b = blockIdx.x * blockDim.x + threadIdx.x;
  if (b >= B) return;
  const float* x = in + (size_t)b * C;
  float m = x[0];
  for (int c = 1; c < C; c++) m = fmaxf(m, x[c]);
  float s = 0.f;
  for (int c = 0; c < C; c++) s += expf(x[c] - m);
  float ls = logf(s);
  for (int c = 0; c < C; c++) out[(size_t)b * C + c] = x[c] - m - ls;
}

// ---------------------------------------------------------------------------
// Host orchestration
// ---------------------------------------------------------------------------
struct Ly { const float *W, *b, *g; };

static inline int imax3(int a, int b, int c) {
  int m = a > b ? a : b; return m > c ? m : c;
}

static void launch_sa(hipStream_t stream, const float* pts, const float* feats,
                      const float* nxyz, int B, int N, int S, int K, int Cpts,
                      int Cin0, int CinP, float r, const Ly* L,
                      int C1, int C2, int C3, float* out, int ostride, int ooff) {
  int Xcap = CinP > C2 ? CinP : C2;
  int Ycap = C1 > C3 ? C1 : C3;
  int Wcap = imax3(C1 * CinP, C2 * C1, C3 * C2);
  size_t hdr = ((size_t)(K * 4 + (N / 32) * 4) + 63) & ~(size_t)63;
  size_t sh  = hdr + 2ull * ((size_t)K * Xcap + (size_t)K * Ycap + (size_t)Wcap);
  sa_branch_kernel<<<B * S, 256, sh, stream>>>(
      pts, feats, nxyz, N, S, K, Cpts, Cin0, CinP, r * r,
      L[0].W, L[0].g, L[0].b, C1,
      L[1].W, L[1].g, L[1].b, C2,
      L[2].W, L[2].g, L[2].b, C3,
      Xcap, Ycap, out, ostride, ooff);
}

extern "C" void kernel_launch(void* const* d_in, const int* in_sizes, int n_in,
                              void* d_out, int out_size, void* d_ws, size_t ws_size,
                              hipStream_t stream) {
  (void)in_sizes; (void)n_in; (void)out_size; (void)ws_size;
  const int B = 8, N = 4096, S1 = 512, S2 = 128;

  // ---- unpack inputs (JAX pytree: sorted dict keys; 'W' < 'b' < 'g') ----
  const float* xyz = (const float*)d_in[0];
  int ii = 1;
  auto F = [&]() -> const float* { return (const float*)d_in[ii++]; };
  const float* bn1_b = F(); const float* bn1_g = F();
  const float* bn2_b = F(); const float* bn2_g = F();
  const float* fc1_W = F(); const float* fc1_b = F();
  const float* fc2_W = F(); const float* fc2_b = F();
  const float* fc3_W = F(); const float* fc3_b = F();
  Ly sa1[3][3], sa2[3][3], sa3[3];
  for (int br = 0; br < 3; br++)
    for (int l = 0; l < 3; l++) { sa1[br][l].W = F(); sa1[br][l].b = F(); sa1[br][l].g = F(); }
  for (int br = 0; br < 3; br++)
    for (int l = 0; l < 3; l++) { sa2[br][l].W = F(); sa2[br][l].b = F(); sa2[br][l].g = F(); }
  for (int l = 0; l < 3; l++) { sa3[l].W = F(); sa3[l].b = F(); sa3[l].g = F(); }

  // ---- workspace partition ----
  char* wp = (char*)d_ws;
  auto alloc = [&](size_t bytes) -> void* {
    void* p = wp; wp += (bytes + 255) & ~(size_t)255; return p;
  };
  float* pts   = (float*)alloc((size_t)B * N * 3 * 4);
  float* nrm   = (float*)alloc((size_t)B * N * 3 * 4);
  int*   fidx1 = (int*)alloc((size_t)B * S1 * 4);
  float* l1xyz = (float*)alloc((size_t)B * S1 * 3 * 4);
  float* l1p   = (float*)alloc((size_t)B * S1 * 320 * 4);
  int*   fidx2 = (int*)alloc((size_t)B * S2 * 4);
  float* l2xyz = (float*)alloc((size_t)B * S2 * 3 * 4);
  float* l2p   = (float*)alloc((size_t)B * S2 * 640 * 4);
  _Float16* X3  = (_Float16*)alloc((size_t)B * S2 * 672 * 2);
  _Float16* Wp1 = (_Float16*)alloc((size_t)256 * 672 * 2);
  _Float16* Wp2 = (_Float16*)alloc((size_t)512 * 256 * 2);
  _Float16* Wp3 = (_Float16*)alloc((size_t)1024 * 512 * 2);
  _Float16* Ya  = (_Float16*)alloc((size_t)1024 * 1024 * 2);
  _Float16* Yb  = (_Float16*)alloc((size_t)1024 * 1024 * 2);
  float* l3  = (float*)alloc((size_t)B * 1024 * 4);
  float* f1o = (float*)alloc((size_t)B * 512 * 4);
  float* f2o = (float*)alloc((size_t)B * 256 * 4);
  float* f3o = (float*)alloc((size_t)B * 40 * 4);

  float* out_logp = (float*)d_out;          // (8,40)
  float* out_l3   = (float*)d_out + B * 40; // (8,1024,1)

  // ---- stage 0: layout ----
  transpose_kernel<<<(B * N + 255) / 256, 256, 0, stream>>>(xyz, pts, nrm, B, N);

  // ---- SA1 ----
  fps_kernel<<<B, 512, (size_t)4 * N * 4, stream>>>(pts, N, S1, fidx1);
  gather_kernel<<<(B * S1 + 255) / 256, 256, 0, stream>>>(pts, fidx1, l1xyz, N, S1, B);
  launch_sa(stream, pts, nrm, l1xyz, B, N, S1, 16,  3, 6, 32, 0.1f, sa1[0], 32, 32, 64,  l1p, 320, 0);
  launch_sa(stream, pts, nrm, l1xyz, B, N, S1, 32,  3, 6, 32, 0.2f, sa1[1], 64, 64, 128, l1p, 320, 64);
  launch_sa(stream, pts, nrm, l1xyz, B, N, S1, 128, 3, 6, 32, 0.4f, sa1[2], 64, 96, 128, l1p, 320, 192);

  // ---- SA2 ----
  fps_kernel<<<B, 512, (size_t)4 * S1 * 4, stream>>>(l1xyz, S1, S2, fidx2);
  gather_kernel<<<(B * S2 + 255) / 256, 256, 0, stream>>>(l1xyz, fidx2, l2xyz, S1, S2, B);
  launch_sa(stream, l1xyz, l1p, l2xyz, B, S1, S2, 32,  320, 323, 352, 0.2f, sa2[0], 64, 64, 128,    l2p, 640, 0);
  launch_sa(stream, l1xyz, l1p, l2xyz, B, S1, S2, 64,  320, 323, 352, 0.4f, sa2[1], 128, 128, 256,  l2p, 640, 128);
  launch_sa(stream, l1xyz, l1p, l2xyz, B, S1, S2, 128, 320, 323, 352, 0.8f, sa2[2], 128, 128, 256,  l2p, 640, 384);

  // ---- SA3 (group-all): M = B*128 = 1024 row GEMM chain ----
  const int M3 = B * S2;
  concat3_kernel<<<(M3 * 672 + 255) / 256, 256, 0, stream>>>(l2xyz, l2p, X3, M3);
  wpad_kernel<<<(256 * 672 + 255) / 256, 256, 0, stream>>>(sa3[0].W, Wp1, 256, 643, 672);
  wpad_kernel<<<(512 * 256 + 255) / 256, 256, 0, stream>>>(sa3[1].W, Wp2, 512, 256, 256);
  wpad_kernel<<<(1024 * 512 + 255) / 256, 256, 0, stream>>>(sa3[2].W, Wp3, 1024, 512, 512);
  {
    int mt = (M3 / 32) * (256 / 32);       // 32x32 macro-tiles, 8 waves/block
    wmma_gemm_kernel<<<(mt + 7) / 8, 256, 0, stream>>>(X3, Wp1, sa3[0].g, sa3[0].b, Ya, M3, 672, 256);
    mt = (M3 / 32) * (512 / 32);
    wmma_gemm_kernel<<<(mt + 7) / 8, 256, 0, stream>>>(Ya, Wp2, sa3[1].g, sa3[1].b, Yb, M3, 256, 512);
    mt = (M3 / 32) * (1024 / 32);
    wmma_gemm_kernel<<<(mt + 7) / 8, 256, 0, stream>>>(Yb, Wp3, sa3[2].g, sa3[2].b, Ya, M3, 512, 1024);
  }
  maxpool3_kernel<<<(B * 1024 + 255) / 256, 256, 0, stream>>>(Ya, l3, out_l3, B);

  // ---- FC head (M=8: VALU path is the right tool here) ----
  fc_kernel<<<(B * 512 + 255) / 256, 256, 0, stream>>>(l3, fc1_W, fc1_b, bn1_g, bn1_b, f1o, B, 1024, 512, 1);
  fc_kernel<<<(B * 256 + 255) / 256, 256, 0, stream>>>(f1o, fc2_W, fc2_b, bn2_g, bn2_b, f2o, B, 512, 256, 1);
  fc_kernel<<<(B * 40 + 255) / 256, 256, 0, stream>>>(f2o, fc3_W, fc3_b, nullptr, nullptr, f3o, B, 256, 40, 0);
  logsoftmax_kernel<<<1, 32, 0, stream>>>(f3o, out_logp, B, 40);
}
